// Model_12945031430952
// MI455X (gfx1250) — matrix operand order repaired and checked by reading, not compiled, not run on hardware
//
#include <hip/hip_runtime.h>
#include <hip/hip_bf16.h>

typedef _Float16 f16;
typedef __attribute__((ext_vector_type(16))) _Float16 v16h;
typedef __attribute__((ext_vector_type(8)))  _Float16 v8h;
typedef __attribute__((ext_vector_type(8)))  float    v8f;

#define B_DIM 8
#define T_DIM 8192
#define CH_MAX 32
#define TT 64          // owned time columns per workgroup
#define TE 96          // extended region (16-col halo each side)
#define HSTR 136       // channel stride (f16 elems) for h LDS tiles (bank-conflict pad)
#define X0STR 40       // channel stride for x0 LDS tile (K padded to 32, +8 pad)
#define FLOW_PACK 106496  // f16 elems of packed weights per flow

#define LOG2E   1.4426950408889634f
#define TWOLOG2E 2.8853900817779268f
#define LN2     0.6931471805599453f

// branch-free fast math (v_exp_f32 / v_rcp_f32 / v_log_f32)
__device__ inline float fast_exp(float x)  { return __builtin_amdgcn_exp2f(x * LOG2E); }
__device__ inline float fast_log(float x)  { return __builtin_amdgcn_logf(x) * LN2; }
__device__ inline float fast_tanh(float x) {
  // tanh(x) = 1 - 2/(e^{2x}+1); saturates correctly at +/-1 for large |x|
  return 1.f - 2.f * __builtin_amdgcn_rcpf(__builtin_amdgcn_exp2f(x * TWOLOG2E) + 1.f);
}

union V16U { v16h v; v8h h[2]; };

__device__ inline v16h frag2(v8h lo, v8h hi) { V16U u; u.h[0] = lo; u.h[1] = hi; return u.v; }

// A fragment: packed fragment-major (lane*16 f16 contiguous) -> 2x b128 global loads
__device__ inline v16h load_A(const f16* __restrict__ p, int lane) {
  const v8h* q = (const v8h*)(p + lane * 16);
  return frag2(q[0], q[1]);
}

// B fragment from LDS [col][ch] layout. contiguous K order used here (matches A packing) for f16 K=32:
// lanes 0-15 hold K=0..15 (VGPR v -> K=2v,2v+1), lanes 16-31 hold K=16..31.
__device__ inline v16h load_B(const f16* __restrict__ hbuf, int stride,
                              int colBase, int chBase, int lane) {
  int col = colBase + (lane & 15);
  col = col < 0 ? 0 : (col > TE - 1 ? TE - 1 : col);   // clamp: halo-only garbage
  int ch = chBase + ((lane < 16) ? 0 : 16);
  const v8h* q = (const v8h*)(hbuf + col * stride + ch);
  return frag2(q[0], q[1]);
}

__device__ inline v8f wmma16(v16h a, v16h b, v8f c) {
  return __builtin_amdgcn_wmma_f32_16x16x32_f16(false, a, false, b, (short)0, c, false, false);
}

// ---------------- weight packing (f32 -> f16, A-fragment lane order) --------
struct FlowW { const float* ws[8]; const float* k0[8]; const float* k1[8]; const float* we[8]; };

__device__ inline int a_frag_K(int lane, int v, int e) {
  // contiguous K order (matches load_B): lanes0-15: v0-7 -> K0..15; lanes16-31: K16..31
  return ((lane < 16) ? 0 : 16) + (v << 1) + e;
}

__global__ __launch_bounds__(256) void pack_kernel(FlowW fw, f16* __restrict__ packW) {
  int k = blockIdx.x;
  int nh = (k < 4) ? 16 : 12;
  f16* base = packW + (size_t)k * FLOW_PACK;
  const float* ws = fw.ws[k];
  const float* c0 = fw.k0[k];
  const float* c1 = fw.k1[k];
  const float* we = fw.we[k];
  for (int idx = threadIdx.x; idx < FLOW_PACK; idx += 256) {
    float val = 0.f;
    if (idx < 4096) {                       // ws: 8 ob blocks, K padded nh->32
      int blk = idx >> 9, r = idx & 511;
      int lane = r >> 4, q = r & 15, v = q >> 1, e = q & 1;
      int M = lane & 15, K = a_frag_K(lane, v, e);
      int o = blk * 16 + M;
      val = (K < nh) ? ws[o * nh + K] : 0.f;
    } else if (idx < 102400) {              // conv layers: (j, ob, kb) blocks
      int idx2 = idx - 4096;
      const float* cc = c0;
      if (idx2 >= 49152) { idx2 -= 49152; cc = c1; }
      int blk = idx2 >> 9, r = idx2 & 511;
      int kb = blk & 3, ob = (blk >> 2) & 7, j = blk >> 5;
      int lane = r >> 4, q = r & 15, v = q >> 1, e = q & 1;
      int M = lane & 15, K = a_frag_K(lane, v, e);
      int o = ob * 16 + M, ii = kb * 32 + K;
      val = cc[(o * 128 + ii) * 3 + j];     // (O,I,KS) row-major
    } else {                                // we: rows padded 2nh->32
      int idx2 = idx - 102400;
      int blk = idx2 >> 9, r = idx2 & 511;
      int kb = blk & 3, ob = blk >> 2;
      int lane = r >> 4, q = r & 15, v = q >> 1, e = q & 1;
      int M = lane & 15, K = a_frag_K(lane, v, e);
      int o = ob * 16 + M;
      val = (o < 2 * nh) ? we[o * 128 + kb * 32 + K] : 0.f;
    }
    base[idx] = (f16)val;
  }
}

// ---------------- fused WN + affine coupling kernel -------------------------
// Each wave owns one output-channel block (ob = wave); A fragments cached in
// registers across the 6 time-block iterations.
__device__ __forceinline__ void conv_stage(const f16* __restrict__ src, f16* __restrict__ dst,
                                           const f16* __restrict__ p, int dil,
                                           int lane, int ob, int t0) {
  v16h a[12];
#pragma unroll
  for (int j = 0; j < 3; ++j)
#pragma unroll
    for (int kb = 0; kb < 4; ++kb)
      a[j * 4 + kb] = load_A(p + (((j * 8 + ob) * 4 + kb) << 9), lane);

  for (int tb = 0; tb < 6; ++tb) {
    int colBase = tb * 16;
    v8f acc = {};
#pragma unroll
    for (int j = 0; j < 3; ++j) {
      int sh = (j - 1) * dil;
#pragma unroll
      for (int kb = 0; kb < 4; ++kb) {
        v16h bf = load_B(src, HSTR, colBase + sh, kb * 32, lane);
        acc = wmma16(a[j * 4 + kb], bf, acc);
      }
    }
    int col = colBase + (lane & 15);
    int gt = t0 + col;
    int ch = ob * 16 + ((lane < 16) ? 0 : 8);       // C/D layout: M = r + (hi?8:0)
    v8h prev = *(const v8h*)(src + col * HSTR + ch);
    bool valid = (gt >= 0) && (gt < T_DIM);
    v8h o;
#pragma unroll
    for (int r = 0; r < 8; ++r) {
      float res = fast_tanh(acc[r]) + (float)prev[r];
      o[r] = valid ? (f16)res : (f16)0.f;
    }
    *(v8h*)(dst + col * HSTR + ch) = o;
  }
}

__global__ __launch_bounds__(256) void wn_kernel(float* __restrict__ x,
    const f16* __restrict__ pk, int nh, float* __restrict__ logs_acc) {
  __shared__ __align__(16) f16 hA[TE * HSTR];
  __shared__ __align__(16) f16 hB[TE * HSTR];
  __shared__ __align__(16) float sunion[TT * 32];   // x0 tile, reused for we-output
  __shared__ float redbuf[8];
  f16* x0s = (f16*)sunion;
  float* outb = sunion;

  const int tid = threadIdx.x, lane = tid & 31, wave = tid >> 5;
  const int b = blockIdx.y;
  const int t0 = (int)blockIdx.x * TT - 16;         // global t of LDS col 0

  const f16* pws = pk;
  const f16* pk0 = pk + 4096;
  const f16* pk1 = pk + 53248;
  const f16* pwe = pk + 102400;

  // load x0 (nh channels) with halo, zero-pad K->32 and out-of-sequence cols
  for (int i = tid; i < TE * 32; i += 256) {
    int col = i % TE, ch = i / TE;
    int gt = t0 + col;
    float v = 0.f;
    if (ch < nh && gt >= 0 && gt < T_DIM) v = x[(b * CH_MAX + ch) * T_DIM + gt];
    x0s[col * X0STR + ch] = (f16)v;
  }
  __syncthreads();

  // stage 0: h0 = tanh(ws @ x0)  (K padded to 32); ob = wave, A cached
  {
    int ob = wave;
    v16h a = load_A(pws + (ob << 9), lane);
    for (int tb = 0; tb < 6; ++tb) {
      int colBase = tb * 16;
      v16h bf = load_B(x0s, X0STR, colBase, 0, lane);
      v8f acc = {};
      acc = wmma16(a, bf, acc);
      int col = colBase + (lane & 15);
      int gt = t0 + col;
      int ch = ob * 16 + ((lane < 16) ? 0 : 8);
      bool valid = (gt >= 0) && (gt < T_DIM);
      v8h o;
#pragma unroll
      for (int r = 0; r < 8; ++r) {
        float res = fast_tanh(acc[r]);
        o[r] = valid ? (f16)res : (f16)0.f;
      }
      *(v8h*)(hA + col * HSTR + ch) = o;
    }
  }
  __syncthreads();

  conv_stage(hA, hB, pk0, 1, lane, wave, t0);  __syncthreads();
  conv_stage(hB, hA, pk1, 2, lane, wave, t0);  __syncthreads();

  // stage 3: out = we @ h  (central TT cols only; 8 tiles, one per wave)
  {
    int ob = wave & 1, tbc = wave >> 1;
    int colBase = 16 + tbc * 16;
    v8f acc = {};
#pragma unroll
    for (int kb = 0; kb < 4; ++kb) {
      v16h a = load_A(pwe + ((ob * 4 + kb) << 9), lane);
      v16h bf = load_B(hA, HSTR, colBase, kb * 32, lane);
      acc = wmma16(a, bf, acc);
    }
    int colc = tbc * 16 + (lane & 15);
    int ch = ob * 16 + ((lane < 16) ? 0 : 8);
    float* op = outb + colc * 32 + ch;
#pragma unroll
    for (int r = 0; r < 8; ++r) op[r] = acc[r];
  }
  __syncthreads();

  // affine coupling: x1 = exp(log_s)*x1 + b   (in place), accumulate sum(log_s)
  float lsum = 0.f;
  for (int i = tid; i < nh * TT; i += 256) {
    int tt = i % TT, ii = i / TT;
    int gt = t0 + 16 + tt;
    float ls = outb[tt * 32 + nh + ii];
    float bv = outb[tt * 32 + ii];
    int xi = (b * CH_MAX + nh + ii) * T_DIM + gt;
    x[xi] = fast_exp(ls) * x[xi] + bv;
    lsum += ls;
  }
#pragma unroll
  for (int s = 16; s > 0; s >>= 1) lsum += __shfl_xor(lsum, s, 32);
  if (lane == 0) redbuf[wave] = lsum;
  __syncthreads();
  if (tid == 0) {
    float s = 0.f;
    for (int w = 0; w < 8; ++w) s += redbuf[w];
    atomicAdd(logs_acc, s);
  }
}

// ---------------- small supporting kernels ----------------------------------
__global__ __launch_bounds__(256) void transpose_kernel(const float* __restrict__ xin,
                                                        float* __restrict__ xout) {
  long idx = (long)blockIdx.x * 256 + threadIdx.x;     // over B*32*T
  if (idx >= (long)B_DIM * CH_MAX * T_DIM) return;
  int t = (int)(idx % T_DIM);
  long r = idx / T_DIM;
  int ch = (int)(r % CH_MAX), b = (int)(r / CH_MAX);
  xout[idx] = xin[((long)b * T_DIM + t) * CH_MAX + ch];
}

__global__ __launch_bounds__(256) void convinv_kernel(const float* __restrict__ src,
    float* __restrict__ dst, const float* __restrict__ W, int c) {
  long idx = (long)blockIdx.x * 256 + threadIdx.x;     // over B*T
  int b = (int)(idx / T_DIM), t = (int)(idx % T_DIM);
  float xin[32];
#pragma unroll
  for (int ch = 0; ch < 32; ++ch)
    xin[ch] = (ch < c) ? src[(b * CH_MAX + ch) * T_DIM + t] : 0.f;
  for (int o = 0; o < c; ++o) {
    float acc = 0.f;
#pragma unroll
    for (int ch = 0; ch < 32; ++ch)
      if (ch < c) acc = fmaf(W[o * c + ch], xin[ch], acc);
    dst[(b * CH_MAX + o) * T_DIM + t] = acc;
  }
}

__global__ __launch_bounds__(256) void copych_kernel(const float* __restrict__ src, int soff,
                                                     float* __restrict__ dst, int doff, int nch) {
  long idx = (long)blockIdx.x * 256 + threadIdx.x;
  if (idx >= (long)B_DIM * nch * T_DIM) return;
  int t = (int)(idx % T_DIM);
  long r = idx / T_DIM;
  int ch = (int)(r % nch), b = (int)(r / nch);
  dst[(b * CH_MAX + doff + ch) * T_DIM + t] = src[(b * CH_MAX + soff + ch) * T_DIM + t];
}

__global__ void zero_kernel(float* p, int n) {
  int i = blockIdx.x * 256 + threadIdx.x;
  if (i < n) p[i] = 0.f;
}

__global__ __launch_bounds__(256) void zsq_kernel(const float* __restrict__ z,
                                                  float* __restrict__ acc) {
  long n = (long)B_DIM * CH_MAX * T_DIM;
  float s = 0.f;
  for (long i = (long)blockIdx.x * blockDim.x + threadIdx.x; i < n;
       i += (long)gridDim.x * blockDim.x) {
    float v = z[i];
    s += v * v;
  }
#pragma unroll
  for (int sh = 16; sh > 0; sh >>= 1) s += __shfl_xor(s, sh, 32);
  if ((threadIdx.x & 31) == 0) atomicAdd(acc, s);
}

// one wave per row: -|a-E|^2/16 logits vs 256 prototypes, log-softmax losses
__global__ __launch_bounds__(256) void irloss_kernel(const float* __restrict__ z,
    const int* __restrict__ yidx, const float* __restrict__ E,
    float* __restrict__ iacc, float* __restrict__ racc) {
  int lane = threadIdx.x & 31, wave = threadIdx.x >> 5;
  int row = blockIdx.x * 8 + wave;
  int b = row / T_DIM, t = row % T_DIM;
#pragma unroll
  for (int mode = 0; mode < 2; ++mode) {
    int chb = (mode == 0) ? (CH_MAX - 16) : 0;         // spk = last 16 ch, pho = first 16
    float av = (lane < 16) ? z[(b * CH_MAX + chb + lane) * T_DIM + t] : 0.f;
    float ad[16];
#pragma unroll
    for (int d = 0; d < 16; ++d) ad[d] = __shfl(av, d, 32);
    float aa = 0.f;
#pragma unroll
    for (int d = 0; d < 16; ++d) aa += ad[d] * ad[d];
    float l[8];
#pragma unroll
    for (int i = 0; i < 8; ++i) {
      int y = i * 32 + lane;
      float dot = 0.f, ee = 0.f;
#pragma unroll
      for (int d = 0; d < 16; ++d) {
        float e = E[y * 16 + d];
        dot = fmaf(ad[d], e, dot);
        ee = fmaf(e, e, ee);
      }
      l[i] = -(aa + ee - 2.f * dot) * 0.0625f;
    }
    float m = l[0];
#pragma unroll
    for (int i = 1; i < 8; ++i) m = fmaxf(m, l[i]);
#pragma unroll
    for (int sh = 16; sh > 0; sh >>= 1) m = fmaxf(m, __shfl_xor(m, sh, 32));
    float se = 0.f;
#pragma unroll
    for (int i = 0; i < 8; ++i) se += fast_exp(l[i] - m);
#pragma unroll
    for (int sh = 16; sh > 0; sh >>= 1) se += __shfl_xor(se, sh, 32);
    float lse = m + fast_log(se);
    if (mode == 0) {
      int label = yidx[row];
      if (lane == (label & 31)) {
        float ll = 0.f;
#pragma unroll
        for (int i = 0; i < 8; ++i)
          if (i == (label >> 5)) ll = l[i];
        atomicAdd(iacc, -(ll - lse));
      }
    } else {
      float sl = 0.f;
#pragma unroll
      for (int i = 0; i < 8; ++i) sl += l[i];
#pragma unroll
      for (int sh = 16; sh > 0; sh >>= 1) sl += __shfl_xor(sl, sh, 32);
      if (lane == 0) atomicAdd(racc, sl - 256.f * lse);
    }
  }
}

struct ConvWs { const float* W[9]; };

__global__ void slogdet_kernel(ConvWs cw, float* __restrict__ acc) {
  __shared__ float mats[9][1024];
  int i = threadIdx.x;
  if (i >= 9) return;
  int n = (i < 4) ? 32 : 24;
  const float* W = cw.W[i];
  float* A = mats[i];
  for (int e = 0; e < n * n; ++e) A[e] = W[e];
  float ld = 0.f;
  for (int col = 0; col < n; ++col) {
    int p = col;
    float best = fabsf(A[col * n + col]);
    for (int r2 = col + 1; r2 < n; ++r2) {
      float v = fabsf(A[r2 * n + col]);
      if (v > best) { best = v; p = r2; }
    }
    if (p != col)
      for (int cc = 0; cc < n; ++cc) {
        float tmp = A[col * n + cc];
        A[col * n + cc] = A[p * n + cc];
        A[p * n + cc] = tmp;
      }
    float piv = A[col * n + col];
    ld += fast_log(fabsf(piv));
    float inv = 1.f / piv;
    for (int r2 = col + 1; r2 < n; ++r2) {
      float f = A[r2 * n + col] * inv;
      for (int cc = col; cc < n; ++cc) A[r2 * n + cc] -= f * A[col * n + cc];
    }
  }
  atomicAdd(acc, (float)B_DIM * (float)T_DIM * ld);
}

__global__ void combine_kernel(const float* __restrict__ sc, float* __restrict__ out) {
  if (threadIdx.x == 0 && blockIdx.x == 0) {
    float denom = (float)B_DIM * (float)CH_MAX * (float)T_DIM;
    float glow = (-sc[0] - sc[1]) / denom;
    float zl = sc[2] / denom;
    float il = sc[3] / ((float)B_DIM * (float)T_DIM);
    float rl = sc[4] / ((float)B_DIM * (float)T_DIM * 256.f);
    out[0] = glow + zl + 10.0f * il + 0.01f * rl;
  }
}

// ---------------- launch ----------------------------------------------------
extern "C" void kernel_launch(void* const* d_in, const int* in_sizes, int n_in,
                              void* d_out, int out_size, void* d_ws, size_t ws_size,
                              hipStream_t stream) {
  (void)in_sizes; (void)n_in; (void)out_size; (void)ws_size;
  const float* x = (const float*)d_in[0];
  const int* yix = (const int*)d_in[1];
  const float* spk = (const float*)d_in[2];
  // d_in[3..11] = convinv_Ws (4x 32x32, 5x 24x24); d_in[12+4k..] = {ws,k0,k1,we} per flow

  char* ws = (char*)d_ws;
  float* scal = (float*)ws;                                  // [0]=logs [1]=logdet [2]=zsq [3]=i [4]=r
  float* bufA = (float*)(ws + 256);
  float* bufB = bufA + (size_t)B_DIM * CH_MAX * T_DIM;
  float* zbuf = bufB + (size_t)B_DIM * CH_MAX * T_DIM;
  f16* packW = (f16*)(zbuf + (size_t)B_DIM * CH_MAX * T_DIM);

  zero_kernel<<<1, 64, 0, stream>>>(scal, 64);

  long total = (long)B_DIM * CH_MAX * T_DIM;
  transpose_kernel<<<(int)((total + 255) / 256), 256, 0, stream>>>(x, bufA);

  FlowW fw;
  for (int k = 0; k < 8; ++k) {
    fw.ws[k] = (const float*)d_in[12 + 4 * k + 0];
    fw.k0[k] = (const float*)d_in[12 + 4 * k + 1];
    fw.k1[k] = (const float*)d_in[12 + 4 * k + 2];
    fw.we[k] = (const float*)d_in[12 + 4 * k + 3];
  }
  pack_kernel<<<8, 256, 0, stream>>>(fw, packW);

  float* cur = bufA;
  float* nxt = bufB;
  int bt_blocks = (B_DIM * T_DIM) / 256;
  for (int k = 0; k < 8; ++k) {
    if (k == 4) {  // early output: z += ch[0:8); x becomes ch[8:32)
      long n8 = (long)B_DIM * 8 * T_DIM;
      copych_kernel<<<(int)((n8 + 255) / 256), 256, 0, stream>>>(cur, 0, zbuf, 0, 8);
      long n24 = (long)B_DIM * 24 * T_DIM;
      copych_kernel<<<(int)((n24 + 255) / 256), 256, 0, stream>>>(cur, 8, nxt, 0, 24);
      float* tw = cur; cur = nxt; nxt = tw;
    }
    int c = (k < 4) ? 32 : 24;
    convinv_kernel<<<bt_blocks, 256, 0, stream>>>(cur, nxt, (const float*)d_in[3 + k], c);
    { float* tw = cur; cur = nxt; nxt = tw; }
    dim3 g(T_DIM / TT, B_DIM);
    wn_kernel<<<g, 256, 0, stream>>>(cur, packW + (size_t)k * FLOW_PACK, c / 2, scal + 0);
  }
  convinv_kernel<<<bt_blocks, 256, 0, stream>>>(cur, nxt, (const float*)d_in[11], 24);
  { float* tw = cur; cur = nxt; nxt = tw; }
  long n24 = (long)B_DIM * 24 * T_DIM;
  copych_kernel<<<(int)((n24 + 255) / 256), 256, 0, stream>>>(cur, 0, zbuf, 8, 24);

  ConvWs cw;
  for (int i = 0; i < 9; ++i) cw.W[i] = (const float*)d_in[3 + i];
  slogdet_kernel<<<1, 32, 0, stream>>>(cw, scal + 1);

  zsq_kernel<<<512, 256, 0, stream>>>(zbuf, scal + 2);
  irloss_kernel<<<(B_DIM * T_DIM) / 8, 256, 0, stream>>>(zbuf, yix, spk, scal + 3, scal + 4);
  combine_kernel<<<1, 1, 0, stream>>>(scal, (float*)d_out);
}
